// Seq2SeqWithAttention_42253888258614
// MI455X (gfx1250) — matrix operand order repaired and checked by reading, not compiled, not run on hardware
//
#include <hip/hip_runtime.h>
#include <cstdint>
#include <cstddef>

typedef _Float16 f16;
typedef _Float16 h8   __attribute__((ext_vector_type(8)));
typedef _Float16 v16h __attribute__((ext_vector_type(16)));
typedef float    v8f  __attribute__((ext_vector_type(8)));
typedef float    flt4 __attribute__((ext_vector_type(4)));

#define BATCH 64
#define SE 512
#define SD 128
#define H  512
#define H4 2048
// packed B tile: 16 kk * 32 lanes * 16 halves = 8192 halves (16KB) per 16-row tile
#define PTILE 8192

#define SHUF16(lo, hh) __builtin_shufflevector(lo, hh, 0,1,2,3,4,5,6,7,8,9,10,11,12,13,14,15)

__device__ __forceinline__ v8f wmma16(v16h a, v16h b, v8f c) {
  return __builtin_amdgcn_wmma_f32_16x16x32_f16(false, a, false, b, (short)0, c, false, false);
}

// A-matrix 16x32 f16 fragment from row-major storage (ISA 7.12.2 16-bit A layout)
__device__ __forceinline__ v16h fragA(const f16* row, int k0, int hi) {
  const f16* p = row + k0 + 8 * hi;
  h8 lo = *(const h8*)(p);
  h8 hh = *(const h8*)(p + 16);
  return SHUF16(lo, hh);
}
// B-matrix fragment from packed (fragment-major) storage: 32 contiguous bytes per lane.
__device__ __forceinline__ v16h fragBp(const f16* base, int off) {
  const f16* p = base + off;
  h8 lo = *(const h8*)(p);
  h8 hh = *(const h8*)(p + 8);
  return SHUF16(lo, hh);
}

__device__ __forceinline__ float sigf(float x) { return 1.f / (1.f + __expf(-x)); }
__device__ __forceinline__ float tanhfast(float x) {
  x = fminf(x, 15.f);
  float e = __expf(2.f * x);
  return (e - 1.f) / (e + 1.f);
}
__device__ __forceinline__ float wred_sum(float v) {
  #pragma unroll
  for (int o = 16; o; o >>= 1) v += __shfl_xor(v, o, 32);
  return v;
}
__device__ __forceinline__ float wred_max(float v) {
  #pragma unroll
  for (int o = 16; o; o >>= 1) v = fmaxf(v, __shfl_xor(v, o, 32));
  return v;
}

// ---------------- prep kernels ----------------
// pack W (rows x 512, row-major f32) into fragment-major f16:
// dst[((nt*16+kk)*32 + L)*16 + h] = W[nt*16 + (L&15)][kk*32 + 8*(L>>4) + (h&7) + 16*(h>>3)]
__global__ void pack_b_kernel(const float* __restrict__ src, f16* __restrict__ dst, int total) {
  int i = blockIdx.x * 256 + threadIdx.x;
  if (i >= total) return;
  int h  = i & 15;
  int L  = (i >> 4) & 31;
  int kk = (i >> 9) & 15;
  int nt = i >> 13;
  int row = nt * 16 + (L & 15);
  int col = kk * 32 + 8 * (L >> 4) + (h & 7) + 16 * (h >> 3);
  dst[i] = (f16)src[(size_t)row * 512 + col];
}
__global__ void addb_kernel(const float* __restrict__ a, const float* __restrict__ b,
                            float* __restrict__ o, int n) {
  int i = blockIdx.x * 256 + threadIdx.x;
  if (i < n) o[i] = a[i] + b[i];
}

// ---------------- encoder: 4 WGs x 512 thr, 16 batches per WG, 2 n-blocks/wave ----------------
__global__ __launch_bounds__(512) void enc_rnn_kernel(
    const float* __restrict__ xin,      // [64][512][1]
    const float* __restrict__ Wih,      // [2048]
    const f16*  __restrict__ WhhP,      // packed, 128 tiles
    const float* __restrict__ bsum,     // [2048]
    f16*  __restrict__ eo16,            // [512][64][512]
    float* __restrict__ hN, float* __restrict__ cN, f16* __restrict__ h16N)
{
  __shared__ __align__(16) f16   h16s[16][H];
  __shared__ __align__(16) float xs[16][SE];
  const int tid = threadIdx.x;
  const int b0  = blockIdx.x * 16;
  for (int i = tid; i < 16 * SE; i += 512) {
    int b = i >> 9, t = i & 511;
    xs[b][t] = xin[(size_t)(b0 + b) * SE + t];
  }
  for (int i = tid; i < 16 * H; i += 512) ((f16*)h16s)[i] = (f16)0.f;
  __syncthreads();

  const int wv = tid >> 5, lane = tid & 31;
  const int lm = lane & 15, hi = lane >> 4;
  const int loff = lane * 16;
  const f16* arow = &h16s[lm][0];

  const f16* bbW[2][4];
  float wih4[2][4], bs4[2][4];
  #pragma unroll
  for (int blk = 0; blk < 2; ++blk) {
    int nb = wv + 16 * blk;
    #pragma unroll
    for (int g = 0; g < 4; ++g) {
      int j = g * H + nb * 16 + lm;
      wih4[blk][g] = Wih[j];
      bs4[blk][g]  = bsum[j];
      bbW[blk][g]  = WhhP + (size_t)(g * 32 + nb) * PTILE;
    }
  }
  float creg[2][8];
  #pragma unroll
  for (int blk = 0; blk < 2; ++blk)
    #pragma unroll
    for (int r = 0; r < 8; ++r) creg[blk][r] = 0.f;

  for (int t = 0; t < SE; ++t) {
    float xv[8];
    #pragma unroll
    for (int r = 0; r < 8; ++r) xv[r] = xs[r + 8 * hi][t];
    v8f acc[2][4];
    #pragma unroll
    for (int blk = 0; blk < 2; ++blk)
      #pragma unroll
      for (int g = 0; g < 4; ++g)
        #pragma unroll
        for (int r = 0; r < 8; ++r) acc[blk][g][r] = xv[r] * wih4[blk][g] + bs4[blk][g];

    // K-loop start rotates with t: defeats LICM hoisting of t-invariant weight loads
    // (same 16 chunks are summed each step, just in rotated order).
    const int kb = t & 15;
    v16h pb0[4], pb1[4];
    #pragma unroll
    for (int g = 0; g < 4; ++g) pb0[g] = fragBp(bbW[0][g], kb * 512 + loff);
    #pragma unroll
    for (int kk = 0; kk < 16; ++kk) {
      const int k  = (kk + kb) & 15;
      const int k1 = (kk + 1 + kb) & 15;
      v16h a = fragA(arow, k * 32, hi);
      #pragma unroll
      for (int g = 0; g < 4; ++g) pb1[g] = fragBp(bbW[1][g], k * 512 + loff);
      #pragma unroll
      for (int g = 0; g < 4; ++g) acc[0][g] = wmma16(a, pb0[g], acc[0][g]);
      if (kk < 15) {
        #pragma unroll
        for (int g = 0; g < 4; ++g) pb0[g] = fragBp(bbW[0][g], k1 * 512 + loff);
      }
      #pragma unroll
      for (int g = 0; g < 4; ++g) acc[1][g] = wmma16(a, pb1[g], acc[1][g]);
    }
    __syncthreads();   // all h16s reads done before overwrite

    f16* eo = eo16 + ((size_t)t * BATCH + b0) * H;
    #pragma unroll
    for (int blk = 0; blk < 2; ++blk) {
      const int n0 = (wv + 16 * blk) * 16;
      #pragma unroll
      for (int r = 0; r < 8; ++r) {
        const int m = r + 8 * hi;
        float iv = sigf(acc[blk][0][r]);
        float fv = sigf(acc[blk][1][r]);
        float gv = tanhfast(acc[blk][2][r]);
        float ov = sigf(acc[blk][3][r]);
        float cn = fv * creg[blk][r] + iv * gv;
        creg[blk][r] = cn;
        float hv = ov * tanhfast(cn);
        f16 hq = (f16)hv;
        h16s[m][n0 + lm] = hq;
        eo[(size_t)m * H + n0 + lm] = hq;
        if (t == SE - 1) {
          hN[(size_t)(b0 + m) * H + n0 + lm]   = hv;
          cN[(size_t)(b0 + m) * H + n0 + lm]   = cn;
          h16N[(size_t)(b0 + m) * H + n0 + lm] = hq;
        }
      }
    }
    __syncthreads();
  }
}

// ---------------- K/V projection: shared-A dual GEMM ----------------
__global__ __launch_bounds__(256) void kv_kernel(
    const f16* __restrict__ eo16,       // [512][64][512]
    const f16* __restrict__ WkP, const f16* __restrict__ WvP,   // packed, 32 tiles each
    const float* __restrict__ attn_b,   // [1536]
    float* __restrict__ K32, float* __restrict__ V32)           // [64][512][512]
{
  const int wv = threadIdx.x >> 5, lane = threadIdx.x & 31;
  const int lm = lane & 15, hi = lane >> 4;
  const int loff = lane * 16;
  const int gw = blockIdx.x * 8 + wv;   // 16384 waves
  const int NT = 2048 * 32;             // paired (K,V) tiles
  for (int tp = gw; tp < NT; tp += 16384) {
    const int mt = tp >> 5;
    const int jt = tp & 31;
    const int j0 = jt * 16;
    const f16* Arow = eo16 + (size_t)(mt * 16 + lm) * H;
    const f16* Bk = WkP + (size_t)jt * PTILE;
    const f16* Bv = WvP + (size_t)jt * PTILE;
    v8f ak, av;
    const float bk = attn_b[512 + j0 + lm], bv = attn_b[1024 + j0 + lm];
    #pragma unroll
    for (int r = 0; r < 8; ++r) { ak[r] = bk; av[r] = bv; }

    v16h aA = fragA(Arow, 0, hi);
    v16h kA = fragBp(Bk, loff), vA = fragBp(Bv, loff);
    #pragma unroll
    for (int kk = 0; kk < 16; kk += 2) {
      v16h aB = fragA(Arow, (kk + 1) * 32, hi);
      v16h kB = fragBp(Bk, (kk + 1) * 512 + loff);
      v16h vB = fragBp(Bv, (kk + 1) * 512 + loff);
      ak = wmma16(aA, kA, ak);
      av = wmma16(aA, vA, av);
      if (kk + 2 < 16) {
        aA = fragA(Arow, (kk + 2) * 32, hi);
        kA = fragBp(Bk, (kk + 2) * 512 + loff);
        vA = fragBp(Bv, (kk + 2) * 512 + loff);
      }
      ak = wmma16(aB, kB, ak);
      av = wmma16(aB, vB, av);
    }
    const int s = mt >> 2;
    #pragma unroll
    for (int r = 0; r < 8; ++r) {
      int b = (mt & 3) * 16 + r + 8 * hi;
      K32[((size_t)b * SE + s) * H + j0 + lm] = ak[r];
      V32[((size_t)b * SE + s) * H + j0 + lm] = av[r];
    }
  }
}

// ---------------- decoder: 16 WGs x 512 thr, 4 batches per WG, all WG-local ----------------
__global__ __launch_bounds__(512) void dec_kernel(
    const float* __restrict__ xin0,     // decoder_inputs [64][128][1] (t=0 only)
    const float* __restrict__ Wih,      // [2048]
    const f16*  __restrict__ WhhP,      // packed, 128 tiles
    const float* __restrict__ bsum,     // [2048]
    const f16*  __restrict__ WqP,       // packed, 32 tiles
    const float* __restrict__ attn_b,   // [1536]
    const f16*  __restrict__ WoP,       // packed, 32 tiles
    const float* __restrict__ attn_ob,  // [512]
    const float* __restrict__ ln_g, const float* __restrict__ ln_b,
    const float* __restrict__ out_w, const float* __restrict__ out_b,
    const float* __restrict__ K32, const float* __restrict__ V32,
    const float* __restrict__ h0, const float* __restrict__ c0,
    const f16*  __restrict__ h16_0,
    float* __restrict__ outp)           // [64][128]
{
  __shared__ __align__(16) f16   h16s[16][H];   // rows 0..3 live, 4..15 zero
  __shared__ __align__(16) f16   cx16[16][H];
  __shared__ __align__(16) float cs[4][H];
  __shared__ __align__(16) float hs[4][H];
  __shared__ __align__(16) float qs[4][H];
  __shared__ __align__(16) float scs[4][SE];
  __shared__ __align__(16) float ys[4][H];
  __shared__ __align__(16) float lngs[H], lnbs[H], ows[H];
  __shared__ float red[4][8];
  __shared__ float xdec[4];

  const int tid = threadIdx.x;
  const int wv = tid >> 5, lane = tid & 31;
  const int lm = lane & 15, hi = lane >> 4;
  const int loff = lane * 16;
  const int b0 = blockIdx.x * 4;

  for (int i = tid; i < 12 * H; i += 512) ((f16*)&h16s[4][0])[i] = (f16)0.f;
  for (int i = tid; i < 16 * H; i += 512) ((f16*)cx16)[i] = (f16)0.f;
  for (int i = tid; i < 4 * H; i += 512) {
    int m = i >> 9, hh = i & 511;
    cs[m][hh]   = c0[(size_t)(b0 + m) * H + hh];
    hs[m][hh]   = h0[(size_t)(b0 + m) * H + hh];
    h16s[m][hh] = h16_0[(size_t)(b0 + m) * H + hh];
  }
  for (int i = tid; i < H; i += 512) { lngs[i] = ln_g[i]; lnbs[i] = ln_b[i]; ows[i] = out_w[i]; }
  if (tid < 4) xdec[tid] = xin0[(size_t)(b0 + tid) * SD];
  __syncthreads();

  const f16* arow = &h16s[lm][0];
  const f16* crow = &cx16[lm][0];
  // cell: blocks wv and wv+16
  const f16* bbW[2][4];
  float wih4[2][4], bs4[2][4];
  #pragma unroll
  for (int blk = 0; blk < 2; ++blk) {
    int nb = wv + 16 * blk;
    #pragma unroll
    for (int g = 0; g < 4; ++g) {
      int j = g * H + nb * 16 + lm;
      wih4[blk][g] = Wih[j];
      bs4[blk][g]  = bsum[j];
      bbW[blk][g]  = WhhP + (size_t)(g * 32 + nb) * PTILE;
    }
  }
  // q/y: tiles 2*wv and 2*wv+1
  const int n0a = (2 * wv) * 16, n0b = (2 * wv + 1) * 16;
  const f16* bbQa = WqP + (size_t)(2 * wv) * PTILE;
  const f16* bbQb = WqP + (size_t)(2 * wv + 1) * PTILE;
  const f16* bbOa = WoP + (size_t)(2 * wv) * PTILE;
  const f16* bbOb = WoP + (size_t)(2 * wv + 1) * PTILE;
  const float bqa = attn_b[n0a + lm], bqb = attn_b[n0b + lm];
  const float boa = attn_ob[n0a + lm], bob = attn_ob[n0b + lm];
  const int bloc = wv >> 2, part = wv & 3;          // 4 waves per batch in attention
  const float* Kb = K32 + (size_t)(b0 + bloc) * SE * H;
  const float* Vb = V32 + (size_t)(b0 + bloc) * SE * H;
  const float scale = 0.044194173824159216f;        // 1/sqrt(512)
  const float outb = out_b[0];

  for (int t = 0; t < SD; ++t) {
    const int kb = t & 15;          // rotated K start: blocks LICM on weight loads
    // ---- P1: LSTM cell, gate-fused, interleaved two-block pipeline ----
    v8f acc[2][4];
    #pragma unroll
    for (int blk = 0; blk < 2; ++blk)
      #pragma unroll
      for (int g = 0; g < 4; ++g)
        #pragma unroll
        for (int r = 0; r < 8; ++r) {
          float xv = (hi == 0 && r < 4) ? xdec[r] : 0.f;
          acc[blk][g][r] = xv * wih4[blk][g] + bs4[blk][g];
        }
    {
      v16h pb0[4], pb1[4];
      #pragma unroll
      for (int g = 0; g < 4; ++g) pb0[g] = fragBp(bbW[0][g], kb * 512 + loff);
      #pragma unroll
      for (int kk = 0; kk < 16; ++kk) {
        const int k  = (kk + kb) & 15;
        const int k1 = (kk + 1 + kb) & 15;
        v16h a = fragA(arow, k * 32, hi);
        #pragma unroll
        for (int g = 0; g < 4; ++g) pb1[g] = fragBp(bbW[1][g], k * 512 + loff);
        #pragma unroll
        for (int g = 0; g < 4; ++g) acc[0][g] = wmma16(a, pb0[g], acc[0][g]);
        if (kk < 15) {
          #pragma unroll
          for (int g = 0; g < 4; ++g) pb0[g] = fragBp(bbW[0][g], k1 * 512 + loff);
        }
        #pragma unroll
        for (int g = 0; g < 4; ++g) acc[1][g] = wmma16(a, pb1[g], acc[1][g]);
      }
    }
    __syncthreads();
    if (hi == 0) {
      #pragma unroll
      for (int blk = 0; blk < 2; ++blk) {
        const int n0 = (wv + 16 * blk) * 16;
        #pragma unroll
        for (int r = 0; r < 4; ++r) {
          float iv = sigf(acc[blk][0][r]);
          float fv = sigf(acc[blk][1][r]);
          float gv = tanhfast(acc[blk][2][r]);
          float ov = sigf(acc[blk][3][r]);
          float cn = fv * cs[r][n0 + lm] + iv * gv;
          cs[r][n0 + lm] = cn;
          float hv = ov * tanhfast(cn);
          hs[r][n0 + lm]   = hv;
          h16s[r][n0 + lm] = (f16)hv;
        }
      }
    }
    __syncthreads();

    // ---- P2: q = h @ Wq^T + bq (2 tiles/wave, rotated, 1-deep prefetch) ----
    {
      v8f qa0, qa1;
      #pragma unroll
      for (int r = 0; r < 8; ++r) { qa0[r] = bqa; qa1[r] = bqb; }
      v16h b0 = fragBp(bbQa, kb * 512 + loff), b1 = fragBp(bbQb, kb * 512 + loff);
      #pragma unroll
      for (int kk = 0; kk < 16; ++kk) {
        const int k  = (kk + kb) & 15;
        const int k1 = (kk + 1 + kb) & 15;
        v16h a = fragA(arow, k * 32, hi);
        v16h n0f, n1f;
        if (kk < 15) { n0f = fragBp(bbQa, k1 * 512 + loff); n1f = fragBp(bbQb, k1 * 512 + loff); }
        qa0 = wmma16(a, b0, qa0);
        qa1 = wmma16(a, b1, qa1);
        b0 = n0f; b1 = n1f;
      }
      if (hi == 0) {
        #pragma unroll
        for (int r = 0; r < 4; ++r) { qs[r][n0a + lm] = qa0[r]; qs[r][n0b + lm] = qa1[r]; }
      }
    }
    __syncthreads();

    // ---- P3: attention (4 waves per batch): scores, softmax, ctx ----
    {
      float a4[4] = {0.f, 0.f, 0.f, 0.f};
      const int sbase = part * 128 + lane;
      const float* K0 = Kb + (size_t)sbase * H;
      for (int h = 0; h < H; h += 4) {
        flt4 qv = *(const flt4*)&qs[bloc][h];
        #pragma unroll
        for (int i = 0; i < 4; ++i) {
          flt4 kv = *(const flt4*)(K0 + (size_t)(32 * i) * H + h);
          a4[i] += qv[0]*kv[0] + qv[1]*kv[1] + qv[2]*kv[2] + qv[3]*kv[3];
        }
      }
      #pragma unroll
      for (int i = 0; i < 4; ++i) a4[i] *= scale;
      float mx = wred_max(fmaxf(fmaxf(a4[0], a4[1]), fmaxf(a4[2], a4[3])));
      if (lane == 0) red[bloc][part] = mx;
      __syncthreads();
      float M = fmaxf(fmaxf(red[bloc][0], red[bloc][1]), fmaxf(red[bloc][2], red[bloc][3]));
      float se = 0.f;
      float ev[4];
      #pragma unroll
      for (int i = 0; i < 4; ++i) {
        ev[i] = __expf(a4[i] - M);
        scs[bloc][sbase + 32 * i] = ev[i];
        se += ev[i];
      }
      se = wred_sum(se);
      if (lane == 0) red[bloc][4 + part] = se;
      __syncthreads();
      float S = red[bloc][4] + red[bloc][5] + red[bloc][6] + red[bloc][7];
      float rinv = 1.f / S;
      float ca[4] = {0.f, 0.f, 0.f, 0.f};
      const float* V0 = Vb + part * 128 + lane;
      #pragma unroll 2
      for (int s = 0; s < SE; ++s) {
        float w = scs[bloc][s];
        const float* vr = V0 + (size_t)s * H;
        __builtin_prefetch(vr + (size_t)8 * H, 0, 0);
        ca[0] += w * vr[0];
        ca[1] += w * vr[32];
        ca[2] += w * vr[64];
        ca[3] += w * vr[96];
      }
      #pragma unroll
      for (int i = 0; i < 4; ++i)
        cx16[bloc][part * 128 + lane + 32 * i] = (f16)(ca[i] * rinv);
    }
    __syncthreads();

    // ---- P4: y = ctx @ Wo^T + bo + h (2 tiles/wave, rotated, 1-deep prefetch) ----
    {
      v8f ya0, ya1;
      #pragma unroll
      for (int r = 0; r < 8; ++r) { ya0[r] = boa; ya1[r] = bob; }
      v16h b0 = fragBp(bbOa, kb * 512 + loff), b1 = fragBp(bbOb, kb * 512 + loff);
      #pragma unroll
      for (int kk = 0; kk < 16; ++kk) {
        const int k  = (kk + kb) & 15;
        const int k1 = (kk + 1 + kb) & 15;
        v16h a = fragA(crow, k * 32, hi);
        v16h n0f, n1f;
        if (kk < 15) { n0f = fragBp(bbOa, k1 * 512 + loff); n1f = fragBp(bbOb, k1 * 512 + loff); }
        ya0 = wmma16(a, b0, ya0);
        ya1 = wmma16(a, b1, ya1);
        b0 = n0f; b1 = n1f;
      }
      if (hi == 0) {
        #pragma unroll
        for (int r = 0; r < 4; ++r) {
          ys[r][n0a + lm] = ya0[r] + hs[r][n0a + lm];
          ys[r][n0b + lm] = ya1[r] + hs[r][n0b + lm];
        }
      }
    }
    __syncthreads();

    // ---- P5: LayerNorm + scalar output (1 wave per batch) ----
    if (wv < 4) {
      float v[16];
      float sm = 0.f;
      #pragma unroll
      for (int i = 0; i < 16; ++i) { v[i] = ys[wv][lane + 32 * i]; sm += v[i]; }
      sm = wred_sum(sm);
      float mu = sm * (1.f / 512.f);
      float sq = 0.f;
      #pragma unroll
      for (int i = 0; i < 16; ++i) { float d = v[i] - mu; sq += d * d; }
      sq = wred_sum(sq);
      float rs = rsqrtf(sq * (1.f / 512.f) + 1e-5f);
      float dt = 0.f;
      #pragma unroll
      for (int i = 0; i < 16; ++i) {
        int hh = lane + 32 * i;
        float yh = (v[i] - mu) * rs * lngs[hh] + lnbs[hh];
        dt += yh * ows[hh];
      }
      dt = wred_sum(dt);
      float o = dt + outb;
      if (lane == 0) {
        outp[(size_t)(b0 + wv) * SD + t] = o;
        xdec[wv] = o;
      }
    }
    __syncthreads();
  }
}

// ---------------- host side ----------------
extern "C" void kernel_launch(void* const* d_in, const int* in_sizes, int n_in,
                              void* d_out, int out_size, void* d_ws, size_t ws_size,
                              hipStream_t stream) {
  const float* enc_in     = (const float*)d_in[0];
  const float* dec_in     = (const float*)d_in[1];
  const float* enc_Wih    = (const float*)d_in[2];
  const float* enc_Whh    = (const float*)d_in[3];
  const float* enc_bih    = (const float*)d_in[4];
  const float* enc_bhh    = (const float*)d_in[5];
  const float* dec_Wih    = (const float*)d_in[6];
  const float* dec_Whh    = (const float*)d_in[7];
  const float* dec_bih    = (const float*)d_in[8];
  const float* dec_bhh    = (const float*)d_in[9];
  const float* attn_in_w  = (const float*)d_in[10];
  const float* attn_in_b  = (const float*)d_in[11];
  const float* attn_out_w = (const float*)d_in[12];
  const float* attn_out_b = (const float*)d_in[13];
  const float* ln_g       = (const float*)d_in[14];
  const float* ln_b       = (const float*)d_in[15];
  const float* out_w      = (const float*)d_in[16];
  const float* out_b      = (const float*)d_in[17];
  float* outp = (float*)d_out;

  char* p = (char*)d_ws;
  auto alloc = [&](size_t bytes) -> void* {
    void* r = (void*)p;
    p += (bytes + 255) & ~(size_t)255;
    return r;
  };
  f16*   encWhhP = (f16*)alloc((size_t)H4 * H * 2);
  f16*   decWhhP = (f16*)alloc((size_t)H4 * H * 2);
  f16*   WqP     = (f16*)alloc((size_t)H * H * 2);
  f16*   WkP     = (f16*)alloc((size_t)H * H * 2);
  f16*   WvP     = (f16*)alloc((size_t)H * H * 2);
  f16*   WoP     = (f16*)alloc((size_t)H * H * 2);
  float* bsum_e  = (float*)alloc((size_t)H4 * 4);
  float* bsum_d  = (float*)alloc((size_t)H4 * 4);
  f16*   eo16    = (f16*)alloc((size_t)SE * BATCH * H * 2);
  float* K32     = (float*)alloc((size_t)BATCH * SE * H * 4);
  float* V32     = (float*)alloc((size_t)BATCH * SE * H * 4);
  float* hN      = (float*)alloc((size_t)BATCH * H * 4);
  float* cN      = (float*)alloc((size_t)BATCH * H * 4);
  f16*   h16N    = (f16*)alloc((size_t)BATCH * H * 2);

  int n;
  n = H4 * H;
  pack_b_kernel<<<(n + 255) / 256, 256, 0, stream>>>(enc_Whh, encWhhP, n);
  pack_b_kernel<<<(n + 255) / 256, 256, 0, stream>>>(dec_Whh, decWhhP, n);
  n = H * H;
  pack_b_kernel<<<(n + 255) / 256, 256, 0, stream>>>(attn_in_w, WqP, n);
  pack_b_kernel<<<(n + 255) / 256, 256, 0, stream>>>(attn_in_w + (size_t)H * H, WkP, n);
  pack_b_kernel<<<(n + 255) / 256, 256, 0, stream>>>(attn_in_w + (size_t)2 * H * H, WvP, n);
  pack_b_kernel<<<(n + 255) / 256, 256, 0, stream>>>(attn_out_w, WoP, n);
  addb_kernel<<<(H4 + 255) / 256, 256, 0, stream>>>(enc_bih, enc_bhh, bsum_e, H4);
  addb_kernel<<<(H4 + 255) / 256, 256, 0, stream>>>(dec_bih, dec_bhh, bsum_d, H4);

  enc_rnn_kernel<<<4, 512, 0, stream>>>(enc_in, enc_Wih, encWhhP, bsum_e,
                                        eo16, hN, cN, h16N);
  kv_kernel<<<2048, 256, 0, stream>>>(eo16, WkP, WvP, attn_in_b, K32, V32);
  dec_kernel<<<16, 512, 0, stream>>>(dec_in, dec_Wih, decWhhP, bsum_d,
                                     WqP, attn_in_b, WoP, attn_out_b,
                                     ln_g, ln_b, out_w, out_b,
                                     K32, V32, hN, cN, h16N, outp);
  (void)in_sizes; (void)n_in; (void)out_size; (void)ws_size;
}
